// XCAttention_69329362092337
// MI455X (gfx1250) — matrix operand order repaired and checked by reading, compiled, not run on hardware
//
#include <hip/hip_runtime.h>

typedef __attribute__((ext_vector_type(16))) __bf16 v16bf;
typedef __attribute__((ext_vector_type(8)))  __bf16 v8bf;
typedef __attribute__((ext_vector_type(8)))  float  v8f;

union FragBF { v16bf v; v8bf h[2]; };

// ---------------------------------------------------------------------------
// fp32 -> bf16 elementwise convert
// ---------------------------------------------------------------------------
__global__ void f32_to_bf16_kernel(const float* __restrict__ in, __bf16* __restrict__ out, int n) {
    int i = blockIdx.x * blockDim.x + threadIdx.x;
    if (i < n) out[i] = (__bf16)in[i];
}

// ---------------------------------------------------------------------------
// Big-tile bf16 WMMA GEMM: C[m,n] = sum_k A[m,k] * B[n,k]  (NT, B row-major)
// 64x32 macro-tile per wave: 8 accumulators (64 VGPRs) + 6 fragments (48)
// stays under the 256 directly-addressable VGPRs (the 64x64 version spilled).
// 12 b128 loads : 8 wmma per K-step = 768 B/wmma per wave.
// Used for the two large GEMMs (QKV and proj): M mult of 64, N mult of 32.
// ---------------------------------------------------------------------------
__global__ void gemm_bf16_wmma_64x32(const __bf16* __restrict__ A, int lda,
                                     const __bf16* __restrict__ B, int ldb,
                                     float* __restrict__ Cf, __bf16* __restrict__ Cb, int ldc,
                                     int M, int N, int K) {
    const int tilesN = N >> 5;
    const int tilesM = M >> 6;
    const int wid = blockIdx.x * (blockDim.x >> 5) + (threadIdx.x >> 5);
    if (wid >= tilesM * tilesN) return;
    const int mt = wid / tilesN, nt = wid % tilesN;
    const int m0 = mt << 6, n0 = nt << 5;
    const int lane = threadIdx.x & 31;
    const int l16 = lane & 15;
    const int hi  = lane >> 4;

    v8f acc[4][2] = {};

    const __bf16* pa[4];
    const __bf16* pb[2];
#pragma unroll
    for (int i = 0; i < 4; i++)
        pa[i] = A + (size_t)(m0 + i * 16 + l16) * lda + (hi ? 8 : 0);
#pragma unroll
    for (int j = 0; j < 2; j++)
        pb[j] = B + (size_t)(n0 + j * 16 + l16) * ldb + (hi ? 8 : 0);

    for (int k0 = 0; k0 < K; k0 += 32) {
        FragBF a[4], b[2];
#pragma unroll
        for (int i = 0; i < 4; i++) {
            a[i].h[0] = *(const v8bf*)(pa[i] + k0);
            a[i].h[1] = *(const v8bf*)(pa[i] + k0 + 16);
        }
#pragma unroll
        for (int j = 0; j < 2; j++) {
            b[j].h[0] = *(const v8bf*)(pb[j] + k0);
            b[j].h[1] = *(const v8bf*)(pb[j] + k0 + 16);
        }
#pragma unroll
        for (int i = 0; i < 4; i++) {
#pragma unroll
            for (int j = 0; j < 2; j++) {
                acc[i][j] = __builtin_amdgcn_wmma_f32_16x16x32_bf16(
                    false, a[i].v, false, b[j].v, (short)0, acc[i][j], false, false);
            }
        }
    }

#pragma unroll
    for (int i = 0; i < 4; i++) {
#pragma unroll
        for (int j = 0; j < 2; j++) {
            const int col = n0 + j * 16 + l16;
#pragma unroll
            for (int r = 0; r < 8; r++) {
                const int row = m0 + i * 16 + hi * 8 + r;
                const size_t idx = (size_t)row * ldc + col;
                if (Cf) Cf[idx] = acc[i][j][r];
                if (Cb) Cb[idx] = (__bf16)acc[i][j][r];
            }
        }
    }
}

// ---------------------------------------------------------------------------
// Generic batched bf16 WMMA GEMM (32x32 tile/wave) for the 96-dim batched
// GEMMs (q^T k and v @ A^T).  Same NT semantics and fragment layouts.
//   A 16x32 bf16: lane<16 holds row m0+l, K {k0..k0+7, k0+16..k0+23};
//                 lane>=16 same row, K {k0+8..15, k0+24..31}  -> two 16B loads
//   B 32x16 bf16: same K split as A, col n0+l          -> two 16B loads
//   D f32: lane<16 -> (M=r, N=lane), lane>=16 -> (M=8+r, N=lane-16)
// ---------------------------------------------------------------------------
__global__ void gemm_bf16_wmma(const __bf16* __restrict__ A, long sAy, long sAz, int lda,
                               const __bf16* __restrict__ B, long sBy, long sBz, int ldb,
                               float* __restrict__ Cf, __bf16* __restrict__ Cb,
                               long sCy, long sCz, int ldc,
                               int M, int N, int K) {
    const int tilesN = N >> 5;
    const int tilesM = M >> 5;
    const int wid = blockIdx.x * (blockDim.x >> 5) + (threadIdx.x >> 5);
    if (wid >= tilesM * tilesN) return;
    const int mt = wid / tilesN, nt = wid % tilesN;
    const int m0 = mt << 5, n0 = nt << 5;
    const int lane = threadIdx.x & 31;
    const int l16 = lane & 15;
    const int hi  = lane >> 4;

    const __bf16* Ab = A + (long)blockIdx.y * sAy + (long)blockIdx.z * sAz;
    const __bf16* Bb = B + (long)blockIdx.y * sBy + (long)blockIdx.z * sBz;

    v8f acc00 = {}, acc01 = {}, acc10 = {}, acc11 = {};

    const __bf16* pa0 = Ab + (size_t)(m0 + l16) * lda + (hi ? 8 : 0);
    const __bf16* pa1 = pa0 + (size_t)16 * lda;
    const __bf16* pb0 = Bb + (size_t)(n0 + l16) * ldb + (hi ? 8 : 0);
    const __bf16* pb1 = pb0 + (size_t)16 * ldb;

    for (int k0 = 0; k0 < K; k0 += 32) {
        FragBF a0, a1, b0, b1;
        a0.h[0] = *(const v8bf*)(pa0 + k0);
        a0.h[1] = *(const v8bf*)(pa0 + k0 + 16);
        a1.h[0] = *(const v8bf*)(pa1 + k0);
        a1.h[1] = *(const v8bf*)(pa1 + k0 + 16);
        b0.h[0] = *(const v8bf*)(pb0 + k0);
        b0.h[1] = *(const v8bf*)(pb0 + k0 + 16);
        b1.h[0] = *(const v8bf*)(pb1 + k0);
        b1.h[1] = *(const v8bf*)(pb1 + k0 + 16);
        acc00 = __builtin_amdgcn_wmma_f32_16x16x32_bf16(false, a0.v, false, b0.v, (short)0, acc00, false, false);
        acc01 = __builtin_amdgcn_wmma_f32_16x16x32_bf16(false, a0.v, false, b1.v, (short)0, acc01, false, false);
        acc10 = __builtin_amdgcn_wmma_f32_16x16x32_bf16(false, a1.v, false, b0.v, (short)0, acc10, false, false);
        acc11 = __builtin_amdgcn_wmma_f32_16x16x32_bf16(false, a1.v, false, b1.v, (short)0, acc11, false, false);
    }

    const size_t cbase = (size_t)((long)blockIdx.y * sCy + (long)blockIdx.z * sCz);
#pragma unroll
    for (int i = 0; i < 2; i++) {
#pragma unroll
        for (int j = 0; j < 2; j++) {
            v8f a = (i == 0) ? (j == 0 ? acc00 : acc01) : (j == 0 ? acc10 : acc11);
            const int col = n0 + j * 16 + l16;
#pragma unroll
            for (int r = 0; r < 8; r++) {
                const int row = m0 + i * 16 + hi * 8 + r;
                const size_t idx = cbase + (size_t)row * ldc + col;
                if (Cf) Cf[idx] = a[r];
                if (Cb) Cb[idx] = (__bf16)a[r];
            }
        }
    }
}

// ---------------------------------------------------------------------------
// Pack/transpose q,k from token-major qkv [16384,2304] to channel-major
// qT/kT [B*h][96][1024] via LDS 32x32 tile.
// grid: x = N/32 (32), y = 96/32 (3), z = s*128 + bh (256); block (32,8)
// ---------------------------------------------------------------------------
__global__ void pack_qk_kernel(const __bf16* __restrict__ qkv,
                               __bf16* __restrict__ qT, __bf16* __restrict__ kT) {
    __shared__ __bf16 tile[32][33];
    const int s  = blockIdx.z >> 7;
    const int bh = blockIdx.z & 127;
    const int b  = bh >> 3, hh = bh & 7;
    const int n0 = blockIdx.x * 32, d0 = blockIdx.y * 32;
    const int tx = threadIdx.x, ty = threadIdx.y;

    const __bf16* src = qkv + (size_t)b * 1024 * 2304 + s * 768 + hh * 96 + d0;
#pragma unroll
    for (int i = 0; i < 4; i++) {
        const int n = n0 + ty + i * 8;
        tile[ty + i * 8][tx] = src[(size_t)n * 2304 + tx];
    }
    __syncthreads();
    __bf16* dst = (s ? kT : qT) + ((size_t)bh * 96 + d0) * 1024 + n0;
#pragma unroll
    for (int i = 0; i < 4; i++) {
        const int dd = ty + i * 8;
        dst[(size_t)dd * 1024 + tx] = tile[tx][dd];
    }
}

// ---------------------------------------------------------------------------
// Per-column (token-dim) inverse L2 norm: scales[r] = 1/max(||col||, 1e-12)
// rows 0..12287 -> q, 12288..24575 -> k. One wave per row of 1024.
// ---------------------------------------------------------------------------
__global__ void col_inv_norm_kernel(const __bf16* __restrict__ qT,
                                    const __bf16* __restrict__ kT,
                                    float* __restrict__ scales) {
    const int wid = blockIdx.x * (blockDim.x >> 5) + (threadIdx.x >> 5);
    const int lane = threadIdx.x & 31;
    if (wid >= 24576) return;
    const __bf16* src = (wid < 12288) ? (qT + (size_t)wid * 1024)
                                      : (kT + (size_t)(wid - 12288) * 1024);
    float s = 0.f;
    for (int i = lane; i < 1024; i += 32) {
        const float v = (float)src[i];
        s += v * v;
    }
#pragma unroll
    for (int o = 16; o; o >>= 1) s += __shfl_down(s, o, 32);
    if (lane == 0) scales[wid] = 1.0f / fmaxf(sqrtf(s), 1e-12f);
}

// ---------------------------------------------------------------------------
// Fused normalize-scale + temperature + softmax over e (96), wave per row.
// logits[dd,e] = Sraw[dd,e] * qs[dd] * ks[e] / temp[h]
// ---------------------------------------------------------------------------
__global__ void softmax_kernel(const float* __restrict__ Sraw,
                               const float* __restrict__ scales,
                               const float* __restrict__ temp,
                               __bf16* __restrict__ Asm) {
    const int wid = blockIdx.x * (blockDim.x >> 5) + (threadIdx.x >> 5);
    const int lane = threadIdx.x & 31;
    if (wid >= 12288) return;
    const int bh = wid / 96, dd = wid % 96;
    const int hh = bh & 7;
    const float invt = 1.0f / temp[hh];
    const float sq = scales[bh * 96 + dd];
    const float* ks = scales + 12288 + bh * 96;
    const float* row = Sraw + (size_t)wid * 96;

    float v[3];
    float mx = -3.4e38f;
#pragma unroll
    for (int j = 0; j < 3; j++) {
        const int e = lane + j * 32;
        v[j] = row[e] * sq * ks[e] * invt;
        mx = fmaxf(mx, v[j]);
    }
#pragma unroll
    for (int o = 16; o; o >>= 1) mx = fmaxf(mx, __shfl_xor(mx, o, 32));
    float sum = 0.f;
#pragma unroll
    for (int j = 0; j < 3; j++) { v[j] = __expf(v[j] - mx); sum += v[j]; }
#pragma unroll
    for (int o = 16; o; o >>= 1) sum += __shfl_xor(sum, o, 32);
    const float inv = 1.0f / sum;
    __bf16* orow = Asm + (size_t)wid * 96;
#pragma unroll
    for (int j = 0; j < 3; j++) orow[lane + j * 32] = (__bf16)(v[j] * inv);
}

// ---------------------------------------------------------------------------
// Fused depthwise 3x3 conv (0.1x) + residual add + LayerNorm. Block per token.
// ---------------------------------------------------------------------------
__global__ void conv_ln_kernel(const float* __restrict__ x,
                               const float* __restrict__ projf,
                               const float* __restrict__ convw,
                               const float* __restrict__ gamma,
                               const float* __restrict__ beta,
                               float* __restrict__ out) {
    const int tok = blockIdx.x;                 // 0..16383
    const int b = tok >> 10, n = tok & 1023;
    const int hy = n >> 5, wx = n & 31;
    const int t = threadIdx.x;                  // 256 threads, 3 channels each
    const float* xb = x + (size_t)b * 1024 * 768;

    float y[3];
#pragma unroll
    for (int j = 0; j < 3; j++) {
        const int c = t + j * 256;
        float acc = 0.f;
#pragma unroll
        for (int di = -1; di <= 1; di++) {
            const int yy = hy + di;
            if (yy < 0 || yy > 31) continue;
#pragma unroll
            for (int dj = -1; dj <= 1; dj++) {
                const int xx = wx + dj;
                if (xx < 0 || xx > 31) continue;
                acc += xb[(size_t)(yy * 32 + xx) * 768 + c] * convw[c * 9 + (di + 1) * 3 + (dj + 1)];
            }
        }
        y[j] = projf[(size_t)tok * 768 + c] + 0.1f * acc;
    }

    float s = y[0] + y[1] + y[2];
    float ss = y[0] * y[0] + y[1] * y[1] + y[2] * y[2];
#pragma unroll
    for (int o = 16; o; o >>= 1) { s += __shfl_down(s, o, 32); ss += __shfl_down(ss, o, 32); }
    __shared__ float sh1[8], sh2[8];
    const int wave = t >> 5, lane = t & 31;
    if (lane == 0) { sh1[wave] = s; sh2[wave] = ss; }
    __syncthreads();
    float S = 0.f, SS = 0.f;
#pragma unroll
    for (int w = 0; w < 8; w++) { S += sh1[w]; SS += sh2[w]; }
    const float mu = S * (1.0f / 768.0f);
    const float var = SS * (1.0f / 768.0f) - mu * mu;
    const float rstd = rsqrtf(var + 1e-5f);
#pragma unroll
    for (int j = 0; j < 3; j++) {
        const int c = t + j * 256;
        out[(size_t)tok * 768 + c] = (y[j] - mu) * rstd * gamma[c] + beta[c];
    }
}

// ---------------------------------------------------------------------------
// Host launch
// ---------------------------------------------------------------------------
extern "C" void kernel_launch(void* const* d_in, const int* in_sizes, int n_in,
                              void* d_out, int out_size, void* d_ws, size_t ws_size,
                              hipStream_t stream) {
    const float* x      = (const float*)d_in[0];   // [16,1024,768]
    const float* w_qkv  = (const float*)d_in[1];   // [2304,768]
    const float* w_proj = (const float*)d_in[2];   // [768,768]
    const float* temp   = (const float*)d_in[3];   // [8,1,1]
    const float* conv_w = (const float*)d_in[4];   // [768,1,3,3]
    const float* gamma  = (const float*)d_in[5];   // [768]
    const float* beta   = (const float*)d_in[6];   // [768]
    float* out = (float*)d_out;

    const int BN = 16 * 1024;      // 16384 tokens
    const int C = 768, C3 = 2304;

    // workspace carve-up (256B aligned)
    char* ws = (char*)d_ws;
    size_t off = 0;
    auto carve = [&](size_t bytes) -> char* {
        char* p = ws + off;
        off = (off + bytes + 255) & ~(size_t)255;
        return p;
    };
    __bf16* xb     = (__bf16*)carve((size_t)BN * C * 2);        // x bf16
    __bf16* wqkvb  = (__bf16*)carve((size_t)C3 * C * 2);        // w_qkv bf16
    __bf16* wprojb = (__bf16*)carve((size_t)C * C * 2);         // w_proj bf16
    __bf16* qkv    = (__bf16*)carve((size_t)BN * C3 * 2);       // qkv bf16 token-major
    __bf16* qT     = (__bf16*)carve((size_t)128 * 96 * 1024 * 2);
    __bf16* kT     = (__bf16*)carve((size_t)128 * 96 * 1024 * 2);
    float*  scales = (float*)carve((size_t)24576 * 4);          // [q|k] inv norms
    float*  Sraw   = (float*)carve((size_t)128 * 96 * 96 * 4);  // raw q^T k
    __bf16* Asm    = (__bf16*)carve((size_t)128 * 96 * 96 * 2); // softmaxed A
    __bf16* xca    = (__bf16*)carve((size_t)BN * C * 2);        // v @ A^T
    float*  projf  = (float*)carve((size_t)BN * C * 4);         // proj output f32

    // 1) converts
    {
        int n = BN * C;
        f32_to_bf16_kernel<<<(n + 255) / 256, 256, 0, stream>>>(x, xb, n);
        n = C3 * C;
        f32_to_bf16_kernel<<<(n + 255) / 256, 256, 0, stream>>>(w_qkv, wqkvb, n);
        n = C * C;
        f32_to_bf16_kernel<<<(n + 255) / 256, 256, 0, stream>>>(w_proj, wprojb, n);
    }

    // 2) QKV GEMM: [16384,768] x [2304,768]^T -> qkv bf16 [16384,2304]
    {
        const int tiles = (BN / 64) * (C3 / 32);                // 18432 waves
        dim3 grid((tiles + 3) / 4, 1, 1);
        gemm_bf16_wmma_64x32<<<grid, 128, 0, stream>>>(
            xb, C, wqkvb, C, nullptr, qkv, C3, BN, C3, C);
    }

    // 3) pack q,k channel-major
    {
        dim3 grid(32, 3, 256), block(32, 8);
        pack_qk_kernel<<<grid, block, 0, stream>>>(qkv, qT, kT);
    }

    // 4) inverse token-dim L2 norms
    col_inv_norm_kernel<<<24576 / 8, 256, 0, stream>>>(qT, kT, scales);

    // 5) Sraw = q^T k per (b,h): batched 96x96x1024
    {
        dim3 grid(3, 8, 16);  // ceil(9 tiles / 4 waves), hh, b
        gemm_bf16_wmma<<<grid, 128, 0, stream>>>(
            qT, (long)96 * 1024, (long)8 * 96 * 1024, 1024,
            kT, (long)96 * 1024, (long)8 * 96 * 1024, 1024,
            Sraw, nullptr, (long)96 * 96, (long)8 * 96 * 96, 96,
            96, 96, 1024);
    }

    // 6) scale + softmax -> Asm bf16
    softmax_kernel<<<12288 / 8, 256, 0, stream>>>(Sraw, scales, temp, Asm);

    // 7) x_xca = v @ A^T per (b,h): batched 1024x96x96 -> xca bf16 [16384,768]
    {
        const int tiles = (1024 / 32) * (96 / 32);              // 96 waves per batch
        dim3 grid((tiles + 3) / 4, 8, 16);
        gemm_bf16_wmma<<<grid, 128, 0, stream>>>(
            qkv + 1536, 96, (long)1024 * 2304, C3,              // v slice of qkv
            Asm, (long)96 * 96, (long)8 * 96 * 96, 96,
            nullptr, xca, 96, (long)1024 * 768, C,
            1024, 96, 96);
    }

    // 8) proj GEMM: [16384,768] x [768,768]^T -> projf f32
    {
        const int tiles = (BN / 64) * (C / 32);                 // 6144 waves
        dim3 grid((tiles + 3) / 4, 1, 1);
        gemm_bf16_wmma_64x32<<<grid, 128, 0, stream>>>(
            xca, C, wprojb, C, projf, nullptr, C, BN, C, C);
    }

    // 9) fused conv + residual + LayerNorm -> out
    conv_ln_kernel<<<BN, 256, 0, stream>>>(x, projf, conv_w, gamma, beta, out);
}
